// ExportBlock_63324997812736
// MI455X (gfx1250) — compile-verified, matrix operand order checked by reading, not run on hardware
//
#include <hip/hip_runtime.h>
#include <math.h>

// ---------------- types ----------------
typedef __bf16 bf16_t;
typedef __attribute__((ext_vector_type(16))) __bf16 v16bf;
typedef __attribute__((ext_vector_type(8)))  __bf16 v8bf;
typedef __attribute__((ext_vector_type(8)))  float  v8f;

// ---------------- model dims ----------------
constexpr int TT      = 1024;   // tokens (B=1)
constexpr int CC      = 2048;   // model dim
constexpr int QKVZ_N  = 8192;   // CONV_DIM + VAL_DIM
constexpr int CONVD   = 6144;   // 2*KEY_DIM + VAL_DIM
constexpr int KEYD    = 2048;
constexpr int VALD    = 2048;
constexpr int HK      = 16;
constexpr int NEXP    = 32;
constexpr int TOPK    = 4;
constexpr int HEXP    = 768;
constexpr int HSH     = 4096;

// ---------------- helpers ----------------
template<int BS>
__device__ inline float block_reduce_sum(float v) {
    __shared__ float sh[BS / 32];
    __shared__ float total;
    int lane = threadIdx.x & 31, w = threadIdx.x >> 5;
    #pragma unroll
    for (int o = 16; o > 0; o >>= 1) v += __shfl_xor(v, o, 32);
    if (lane == 0) sh[w] = v;
    __syncthreads();
    if (w == 0) {
        float t = (threadIdx.x < BS / 32) ? sh[threadIdx.x] : 0.f;
        #pragma unroll
        for (int o = BS / 64; o > 0; o >>= 1) t += __shfl_xor(t, o, 32);
        if (lane == 0) total = t;
    }
    __syncthreads();
    return total;
}

__device__ inline float sigmoidf_(float x) { return 1.f / (1.f + expf(-x)); }
__device__ inline float siluf_(float x)    { return x / (1.f + expf(-x)); }

// ================= bf16 WMMA GEMM =================
// C[M,N] = epilogue( A[M,K](bf16,row) @ B[N,K](fp32,row)^T )
// Block tile 64x128, 8 waves (2x4), wave tile 32x32 (2x2 wmma frags), K slice 64.
// A tile is staged via CDNA5 async global->LDS copies (ASYNCcnt); B is loaded
// fp32, converted to bf16 in-register, and stored to LDS.
constexpr int BM = 64, BN = 128, BK = 64, LDA_S = 72, LDB_S = 72;

__device__ inline v16bf load_frag_lds(const bf16_t* rowbase, int lane) {
    int kbase = (lane < 16) ? 0 : 8;   // ISA 16-bit operand K striping
    v8bf lo = *(const v8bf*)(rowbase + kbase);
    v8bf hi = *(const v8bf*)(rowbase + 16 + kbase);
    return __builtin_shufflevector(lo, hi, 0,1,2,3,4,5,6,7,8,9,10,11,12,13,14,15);
}

__device__ inline void async_copy_b128(unsigned lds_off, const void* gaddr) {
    asm volatile("global_load_async_to_lds_b128 %0, %1, off"
                 :: "v"(lds_off), "v"(gaddr) : "memory");
}
__device__ inline void wait_asynccnt0() {
    asm volatile("s_wait_asynccnt 0x0" ::: "memory");
}

__global__ __launch_bounds__(256)
void gemm_bf16_kernel(const bf16_t* __restrict__ A, const float* __restrict__ B,
                      float* __restrict__ C, int M, int N, int K,
                      const float* __restrict__ resid,
                      const float* __restrict__ rowscale, int rsStride,
                      int accumulate) {
    __shared__ bf16_t As[BM * LDA_S];
    __shared__ bf16_t Bs[BN * LDB_S];

    const int tid  = threadIdx.x;
    const int lane = tid & 31;
    const int wave = tid >> 5;
    const int wm   = wave >> 2;          // 0..1
    const int wn   = wave & 3;           // 0..3
    const int row0 = blockIdx.y * BM;
    const int col0 = blockIdx.x * BN;

    v8f acc[2][2] = {};

    // staging maps
    const int ar = tid >> 2, ak = (tid & 3) * 16;  // A: 16 bf16 / thread (2 async b128)
    const int br = tid >> 1, bk = (tid & 1) * 32;  // B: 32 fp32 / thread

    const unsigned a_lds0 = (unsigned)(unsigned long long)(&As[ar * LDA_S + ak]);
    const unsigned a_lds1 = (unsigned)(unsigned long long)(&As[ar * LDA_S + ak + 8]);

    for (int k0 = 0; k0 < K; k0 += BK) {
        __syncthreads();
        // A tile: bf16 global -> LDS via async DMA (no VGPR round-trip)
        {
            const bf16_t* ap = A + (size_t)(row0 + ar) * K + k0 + ak;
            async_copy_b128(a_lds0, ap);
            async_copy_b128(a_lds1, ap + 8);
        }
        // B tile: fp32 global -> convert -> LDS (32 bf16 per thread)
        {
            alignas(16) bf16_t tmp[32];
            const float* bp = B + (size_t)(col0 + br) * K + k0 + bk;
            #pragma unroll
            for (int q = 0; q < 8; ++q) {
                float4 f = *(const float4*)(bp + 4 * q);
                tmp[4*q+0] = (bf16_t)f.x; tmp[4*q+1] = (bf16_t)f.y;
                tmp[4*q+2] = (bf16_t)f.z; tmp[4*q+3] = (bf16_t)f.w;
            }
            #pragma unroll
            for (int q = 0; q < 4; ++q)
                *(uint4*)(&Bs[br * LDB_S + bk + 8 * q]) = *(const uint4*)(&tmp[8 * q]);
            if (k0 + BK < K)                    // CDNA5 global_prefetch_b8 for next slice
                __builtin_prefetch(bp + BK, 0, 1);
        }
        wait_asynccnt0();                       // own async A copies landed in LDS
        __syncthreads();

        #pragma unroll
        for (int ks = 0; ks < BK; ks += 32) {
            v16bf af[2], bf[2];
            #pragma unroll
            for (int mt = 0; mt < 2; ++mt) {
                int m = wm * 32 + mt * 16 + (lane & 15);
                af[mt] = load_frag_lds(&As[m * LDA_S + ks], lane);
            }
            #pragma unroll
            for (int nt = 0; nt < 2; ++nt) {
                int n = wn * 32 + nt * 16 + (lane & 15);
                bf[nt] = load_frag_lds(&Bs[n * LDB_S + ks], lane);
            }
            #pragma unroll
            for (int mt = 0; mt < 2; ++mt)
                #pragma unroll
                for (int nt = 0; nt < 2; ++nt)
                    acc[mt][nt] = __builtin_amdgcn_wmma_f32_16x16x32_bf16(
                        false, af[mt], false, bf[nt], (short)0, acc[mt][nt], false, false);
        }
    }

    // epilogue: C = acc*rowscale + resid + (accumulate ? C : 0)
    const int mofs = (lane < 16) ? 0 : 8;
    const int ncol = col0 + wn * 32 + (lane & 15);
    #pragma unroll
    for (int mt = 0; mt < 2; ++mt) {
        #pragma unroll
        for (int nt = 0; nt < 2; ++nt) {
            #pragma unroll
            for (int r = 0; r < 8; ++r) {
                int mg = row0 + wm * 32 + mt * 16 + mofs + r;
                size_t idx = (size_t)mg * N + ncol + nt * 16;
                float v = acc[mt][nt][r];
                if (rowscale)   v *= rowscale[(size_t)mg * rsStride];
                if (resid)      v += resid[idx];
                if (accumulate) v += C[idx];
                C[idx] = v;
            }
        }
    }
}

// ================= elementwise / small kernels =================
__global__ void rmsnorm_bf16_kernel(const float* __restrict__ x, const float* __restrict__ w,
                                    bf16_t* __restrict__ out) {
    int t = blockIdx.x;
    float ss = 0.f;
    for (int i = threadIdx.x; i < CC; i += 256) { float v = x[(size_t)t*CC + i]; ss += v*v; }
    float tot = block_reduce_sum<256>(ss);
    float s = rsqrtf(tot / CC + 1e-6f);
    for (int i = threadIdx.x; i < CC; i += 256)
        out[(size_t)t*CC + i] = (bf16_t)(x[(size_t)t*CC + i] * s * w[i]);
}

__global__ void ab_proj_kernel(const bf16_t* __restrict__ h, const float* __restrict__ wb,
                               const float* __restrict__ wa, float* __restrict__ beta,
                               float* __restrict__ alpha) {
    int t = blockIdx.x, g = threadIdx.x >> 3, l = threadIdx.x & 7;
    const float* wrow = (g < 16) ? (wb + (size_t)g * CC) : (wa + (size_t)(g-16) * CC);
    float p = 0.f;
    for (int j = l; j < CC; j += 8) p += (float)h[(size_t)t*CC + j] * wrow[j];
    #pragma unroll
    for (int o = 4; o > 0; o >>= 1) p += __shfl_xor(p, o, 32);
    if (l == 0) {
        float v = sigmoidf_(p);
        if (g < 16) beta[t*16 + g] = v; else alpha[t*16 + (g-16)] = v;
    }
}

__global__ void conv_silu_kernel(const float* __restrict__ qkvz, const float* __restrict__ cw,
                                 const float* __restrict__ cstate, const int* __restrict__ ipos,
                                 float* __restrict__ y) {
    int c = blockIdx.x * 256 + threadIdx.x;
    int t = blockIdx.y;
    float keep = (ipos[0] == 0) ? 0.f : 1.f;
    float acc = 0.f;
    #pragma unroll
    for (int j = 0; j < 4; ++j) {
        int tp = t - 3 + j;
        float v = (tp >= 0) ? qkvz[(size_t)tp * QKVZ_N + c] : keep * cstate[(size_t)c*4 + (tp + 4)];
        acc += cw[(size_t)c*4 + j] * v;
    }
    y[(size_t)t * CONVD + c] = siluf_(acc);
}

__global__ void l2norm_kernel(float* __restrict__ y) {
    int t = blockIdx.x, hh = blockIdx.y;             // hh in [0,32): q heads then k heads
    size_t base = (size_t)t * CONVD + (size_t)hh * 128;
    float v = y[base + threadIdx.x];
    float ss = block_reduce_sum<128>(v * v);
    y[base + threadIdx.x] = v / fmaxf(sqrtf(ss), 1e-12f);
}

// GDN recurrent scan: one workgroup per head; 128x128 fp32 state in LDS.
__global__ __launch_bounds__(256)
void gdn_scan_kernel(const float* __restrict__ y, const float* __restrict__ alpha,
                     const float* __restrict__ beta, const float* __restrict__ rstate,
                     const int* __restrict__ ipos, float* __restrict__ attn) {
    extern __shared__ float smem[];
    float* S   = smem;            // 128*128
    float* kk  = S + 16384;       // 128
    float* vv  = kk + 128;        // 128
    float* qq  = vv + 128;        // 128
    float* pvt = qq + 128;        // 256
    const int head = blockIdx.x;
    const int tid  = threadIdx.x;
    const float keep = (ipos[0] == 0) ? 0.f : 1.f;

    for (int i = tid; i < 16384; i += 256)
        S[i] = keep * rstate[(size_t)head * 16384 + i];
    __syncthreads();

    const int vcol  = tid & 127;
    const int khalf = tid >> 7;
    for (int t = 0; t < TT; ++t) {
        if (tid < 128) {
            size_t rb = (size_t)t * CONVD;
            qq[tid] = y[rb + (size_t)head * 128 + tid];
            kk[tid] = y[rb + KEYD + (size_t)head * 128 + tid];
            vv[tid] = y[rb + 2 * KEYD + (size_t)head * 128 + tid];
        }
        float a = alpha[t * 16 + head];
        float b = beta[t * 16 + head];
        __syncthreads();
        float p = 0.f, vval = vv[vcol];
        #pragma unroll 8
        for (int ki = 0; ki < 64; ++ki) {
            int k = khalf * 64 + ki;
            float s = S[k * 128 + vcol];
            s = a * s + b * kk[k] * vval;
            S[k * 128 + vcol] = s;
            p += qq[k] * s;
        }
        pvt[tid] = p;
        __syncthreads();
        if (tid < 128)
            attn[(size_t)t * VALD + (size_t)head * 128 + tid] = pvt[tid] + pvt[tid + 128];
        __syncthreads();
    }
}

__global__ void attn_gate_norm_kernel(const float* __restrict__ attn, const float* __restrict__ qkvz,
                                      const float* __restrict__ gw, bf16_t* __restrict__ out) {
    int t = blockIdx.x, h = blockIdx.y, i = threadIdx.x;
    size_t idx = (size_t)t * VALD + (size_t)h * 128 + i;
    float o = attn[idx];
    float tot = block_reduce_sum<128>(o * o);
    float r = rsqrtf(tot / 128.f + 1e-6f);
    float z = qkvz[(size_t)t * QKVZ_N + CONVD + (size_t)h * 128 + i];
    out[idx] = (bf16_t)(o * r * gw[i] * sigmoidf_(z));
}

__global__ void router_kernel(const bf16_t* __restrict__ h2, const float* __restrict__ gw,
                              float* __restrict__ coeff) {
    __shared__ float sc[NEXP];
    int t = blockIdx.x, g = threadIdx.x >> 3, l = threadIdx.x & 7;
    float p = 0.f;
    for (int j = l; j < CC; j += 8) p += (float)h2[(size_t)t*CC + j] * gw[(size_t)g*CC + j];
    #pragma unroll
    for (int o = 4; o > 0; o >>= 1) p += __shfl_xor(p, o, 32);
    if (l == 0) sc[g] = p;
    __syncthreads();
    if (threadIdx.x == 0) {
        float s[NEXP], cf[NEXP], tw[TOPK]; int ti[TOPK];
        for (int e = 0; e < NEXP; ++e) { s[e] = sc[e]; cf[e] = 0.f; }
        for (int j = 0; j < TOPK; ++j) {
            int bi = 0; float bv = s[0];
            for (int e = 1; e < NEXP; ++e) if (s[e] > bv) { bv = s[e]; bi = e; }
            tw[j] = bv; ti[j] = bi; s[bi] = -1e30f;
        }
        float m = tw[0], den = 0.f;
        for (int j = 0; j < TOPK; ++j) { tw[j] = expf(tw[j] - m); den += tw[j]; }
        for (int j = 0; j < TOPK; ++j) cf[ti[j]] += tw[j] / den;
        for (int e = 0; e < NEXP; ++e) coeff[(size_t)t * NEXP + e] = cf[e];
    }
}

__global__ void fill_zero_kernel(float* __restrict__ p, int n) {
    int i = blockIdx.x * 256 + threadIdx.x;
    if (i < n) p[i] = 0.f;
}

__global__ void silu_mul_bf16_kernel(const float* __restrict__ g, const float* __restrict__ u,
                                     bf16_t* __restrict__ o, int n) {
    int i = blockIdx.x * 256 + threadIdx.x;
    if (i < n) o[i] = (bf16_t)(siluf_(g[i]) * u[i]);
}

__global__ void sgate_kernel(const bf16_t* __restrict__ h2, const float* __restrict__ w,
                             float* __restrict__ sg) {
    int t = blockIdx.x;
    float p = 0.f;
    for (int j = threadIdx.x; j < CC; j += 128) p += (float)h2[(size_t)t*CC + j] * w[j];
    float tot = block_reduce_sum<128>(p);
    if (threadIdx.x == 0) sg[t] = sigmoidf_(tot);
}

__global__ void final_combine_kernel(const float* __restrict__ x1, const float* __restrict__ routed,
                                     const float* __restrict__ sh, const float* __restrict__ sg,
                                     float* __restrict__ out, int n) {
    int i = blockIdx.x * 256 + threadIdx.x;
    if (i < n) { int t = i >> 11; out[i] = x1[i] + routed[i] + sg[t] * sh[i]; }
}

// ================= host =================
static inline void run_gemm(const bf16_t* A, const float* B, float* C, int M, int N, int K,
                            const float* resid, const float* rowscale, int rsStride,
                            int accumulate, hipStream_t s) {
    dim3 g(N / BN, M / BM);
    gemm_bf16_kernel<<<g, 256, 0, s>>>(A, B, C, M, N, K, resid, rowscale, rsStride, accumulate);
}

extern "C" void kernel_launch(void* const* d_in, const int* in_sizes, int n_in,
                              void* d_out, int out_size, void* d_ws, size_t ws_size,
                              hipStream_t stream) {
    (void)in_sizes; (void)n_in; (void)out_size; (void)ws_size;
    const float* x      = (const float*)d_in[0];
    const int*   ipos   = (const int*)  d_in[1];
    const float* ln1    = (const float*)d_in[2];
    const float* ln2    = (const float*)d_in[3];
    const float* wqkvz  = (const float*)d_in[4];
    const float* wb     = (const float*)d_in[5];
    const float* wa     = (const float*)d_in[6];
    const float* convw  = (const float*)d_in[7];
    const float* cstate = (const float*)d_in[8];
    const float* rstate = (const float*)d_in[9];
    const float* gdnw   = (const float*)d_in[10];
    const float* wout   = (const float*)d_in[11];
    const float* gatew  = (const float*)d_in[12];
    const float* wegate = (const float*)d_in[13];
    const float* weup   = (const float*)d_in[14];
    const float* wedown = (const float*)d_in[15];
    const float* swgate = (const float*)d_in[16];
    const float* swup   = (const float*)d_in[17];
    const float* swdown = (const float*)d_in[18];
    const float* shgw   = (const float*)d_in[19];

    char* w = (char*)d_ws;
    // workspace layout (bytes)
    const size_t OFF_HBF   = 0;                               // 1024*2048 bf16  = 4 MB
    const size_t OFF_QKVZ  = OFF_HBF   + (size_t)TT*CC*2;     // 1024*8192 f32   = 32 MB
    const size_t OFF_Y     = OFF_QKVZ  + (size_t)TT*QKVZ_N*4; // 1024*6144 f32   = 24 MB
    const size_t OFF_ALPHA = OFF_Y     + (size_t)TT*CONVD*4;
    const size_t OFF_BETA  = OFF_ALPHA + (size_t)TT*16*4;
    const size_t OFF_ATTN  = OFF_BETA  + (size_t)TT*16*4;     // 8 MB
    const size_t OFF_ATTNG = OFF_ATTN  + (size_t)TT*VALD*4;   // bf16 4 MB
    const size_t OFF_X1    = OFF_ATTNG + (size_t)TT*VALD*2;   // 8 MB
    const size_t OFF_H2BF  = OFF_X1    + (size_t)TT*CC*4;     // bf16 4 MB
    const size_t OFF_COEFF = OFF_H2BF  + (size_t)TT*CC*2;
    const size_t OFF_STMP  = OFF_COEFF + (size_t)TT*NEXP*4;   // bf16 8 MB
    const size_t OFF_SGATE = OFF_STMP  + (size_t)TT*HSH*2;
    // aliases (buffers whose lifetimes ended)
    const size_t OFF_SG    = OFF_QKVZ;                        // 16 MB (qkvz free after attn-gate)
    const size_t OFF_SU    = OFF_QKVZ + (size_t)TT*HSH*4;
    const size_t OFF_ROUT  = OFF_Y;                           // 8 MB  (y free after scan)
    const size_t OFF_HG    = OFF_Y + (8u<<20);
    const size_t OFF_HU    = OFF_HG + (size_t)TT*HEXP*4;
    const size_t OFF_HID   = OFF_HU + (size_t)TT*HEXP*4;
    const size_t OFF_SHOUT = OFF_Y + (16u<<20);

    bf16_t* hbf    = (bf16_t*)(w + OFF_HBF);
    float*  qkvz   = (float*) (w + OFF_QKVZ);
    float*  ybuf   = (float*) (w + OFF_Y);
    float*  alpha  = (float*) (w + OFF_ALPHA);
    float*  beta   = (float*) (w + OFF_BETA);
    float*  attn   = (float*) (w + OFF_ATTN);
    bf16_t* attng  = (bf16_t*)(w + OFF_ATTNG);
    float*  x1     = (float*) (w + OFF_X1);
    bf16_t* h2bf   = (bf16_t*)(w + OFF_H2BF);
    float*  coeff  = (float*) (w + OFF_COEFF);
    bf16_t* stmp   = (bf16_t*)(w + OFF_STMP);
    float*  sgv    = (float*) (w + OFF_SGATE);
    float*  sg     = (float*) (w + OFF_SG);
    float*  su     = (float*) (w + OFF_SU);
    float*  routed = (float*) (w + OFF_ROUT);
    float*  hg     = (float*) (w + OFF_HG);
    float*  hu     = (float*) (w + OFF_HU);
    bf16_t* hid    = (bf16_t*)(w + OFF_HID);
    float*  shout  = (float*) (w + OFF_SHOUT);

    // 1. h = rms(x, ln1) -> bf16
    rmsnorm_bf16_kernel<<<TT, 256, 0, stream>>>(x, ln1, hbf);
    // 2. qkvz = h @ Wqkvz^T
    run_gemm(hbf, wqkvz, qkvz, TT, QKVZ_N, CC, nullptr, nullptr, 0, 0, stream);
    // 3. alpha/beta
    ab_proj_kernel<<<TT, 256, 0, stream>>>(hbf, wb, wa, beta, alpha);
    // 4. depthwise conv + silu
    conv_silu_kernel<<<dim3(CONVD/256, TT), 256, 0, stream>>>(qkvz, convw, cstate, ipos, ybuf);
    // 5. l2norm q and k (32 head-slices)
    l2norm_kernel<<<dim3(TT, 32), 128, 0, stream>>>(ybuf);
    // 6. gated delta-rule scan (state in LDS, one WGP per head)
    gdn_scan_kernel<<<HK, 256, 17024 * sizeof(float), stream>>>(ybuf, alpha, beta, rstate, ipos, attn);
    // 7. per-head RMS * sigmoid(z) -> bf16
    attn_gate_norm_kernel<<<dim3(TT, HK), 128, 0, stream>>>(attn, qkvz, gdnw, attng);
    // 8. x1 = x + attng @ Wout^T
    run_gemm(attng, wout, x1, TT, CC, VALD, x, nullptr, 0, 0, stream);
    // 9. h2 = rms(x1, ln2) -> bf16
    rmsnorm_bf16_kernel<<<TT, 256, 0, stream>>>(x1, ln2, h2bf);
    // 10. router -> dense coeff
    router_kernel<<<TT, 256, 0, stream>>>(h2bf, gatew, coeff);
    // 11. routed = 0
    fill_zero_kernel<<<(TT*CC + 255)/256, 256, 0, stream>>>(routed, TT*CC);
    // 12. dense coeff-weighted experts (weight streaming-bound; coeff scaling in epilogue)
    for (int e = 0; e < NEXP; ++e) {
        const float* wg = wegate + (size_t)e * HEXP * CC;
        const float* wu = weup   + (size_t)e * HEXP * CC;
        const float* wd = wedown + (size_t)e * CC * HEXP;
        run_gemm(h2bf, wg, hg, TT, HEXP, CC, nullptr, nullptr, 0, 0, stream);
        run_gemm(h2bf, wu, hu, TT, HEXP, CC, nullptr, nullptr, 0, 0, stream);
        silu_mul_bf16_kernel<<<(TT*HEXP + 255)/256, 256, 0, stream>>>(hg, hu, hid, TT*HEXP);
        run_gemm(hid, wd, routed, TT, CC, HEXP, nullptr, coeff + e, NEXP, 1, stream);
    }
    // 13. shared MLP
    run_gemm(h2bf, swgate, sg, TT, HSH, CC, nullptr, nullptr, 0, 0, stream);
    run_gemm(h2bf, swup,   su, TT, HSH, CC, nullptr, nullptr, 0, 0, stream);
    silu_mul_bf16_kernel<<<(TT*HSH + 255)/256, 256, 0, stream>>>(sg, su, stmp, TT*HSH);
    run_gemm(stmp, swdown, shout, TT, CC, HSH, nullptr, nullptr, 0, 0, stream);
    // 14. shared gate
    sgate_kernel<<<TT, 128, 0, stream>>>(h2bf, shgw, sgv);
    // 15. out = x1 + routed + sgate*shared
    final_combine_kernel<<<(TT*CC + 255)/256, 256, 0, stream>>>(x1, routed, shout, sgv,
                                                                (float*)d_out, TT*CC);
}
